// SA_19344532701645
// MI455X (gfx1250) — matrix operand order repaired and checked by reading, not compiled, not run on hardware
//
#include <hip/hip_runtime.h>
#include <hip/hip_bf16.h>

// Problem constants (reference: B,S,D = 2,512,128 ; DG=64, NG=2)
#define B_  2
#define S_  512
#define D_  128
#define DG_ 64
#define NG_ 2

typedef __attribute__((ext_vector_type(16))) __bf16 v16bf;
typedef __attribute__((ext_vector_type(8)))  float  v8f;

__device__ __forceinline__ unsigned short f2bfu(float f) {
  union { float f; unsigned int u; } v; v.f = f;
  // round-to-nearest-even bf16 (setup paths only)
  unsigned int r = (v.u + 0x7FFFu + ((v.u >> 16) & 1u)) >> 16;
  return (unsigned short)r;
}

// Single-instruction ReLU: v_med3_f32(x, 0, BIG)
__device__ __forceinline__ float reluf(float x) {
  return __builtin_amdgcn_fmed3f(x, 0.f, 3.0e38f);
}

// Swap 16-lane rows within the wave32 (equivalent to shfl_xor 16), pure VALU.
__device__ __forceinline__ float xrow_swap(float x) {
  union { float f; int i; } a, r;
  a.f = x;
  r.i = __builtin_amdgcn_permlanex16(a.i, a.i, 0x76543210, 0xfedcba98,
                                     false, false);
  return r.f;
}

// Packed bf16: relu(q + nk) on two elements per dword, pure CDNA5 VOP3P.
__device__ __forceinline__ unsigned int pk_relu_add(unsigned int qv, unsigned int nkv) {
  unsigned int s, r;
  asm("v_pk_add_bf16 %0, %1, %2" : "=v"(s) : "v"(qv), "v"(nkv));
  asm("v_pk_max_num_bf16 %0, %1, 0" : "=v"(r) : "v"(s));
  return r;
}

// ---------------------------------------------------------------------------
// Kernel 0: pre-pack W2^T into the WMMA A-fragment lane layout (bf16), once.
// A layout (16-bit A 16x32): lane<16 -> row n=lane, K=0..15;
// lane>=16 -> row n=lane-16, K=16..31 (matches B).  A[n,K] = W2[K,n].
// afw[lane][chunk][j], 32 lanes * 32 bf16 = 2 KB.
// ---------------------------------------------------------------------------
__global__ __launch_bounds__(32)
void SA_w2_kernel(const float* __restrict__ W2,       // [64,16]
                  unsigned short* __restrict__ afw) { // bf16 [32][2][16]
  const int lane = threadIdx.x;
  const int r15 = lane & 15, half = lane >> 4;
  #pragma unroll
  for (int c = 0; c < 2; ++c)
    #pragma unroll
    for (int j = 0; j < 16; ++j) {
      const int kk = c * 32 + half * 16 + j;
      afw[lane * 32 + c * 16 + j] = f2bfu(W2[kk * 16 + r15]);
    }
}

// ---------------------------------------------------------------------------
// Kernel 1: QKV projection + folded layer-1.
//   qw1  = bf16(Q@W1 + b1)   (per group)
//   nkw1 = bf16(-(K@W1))     (negated, so inner loop is pk_add)
// Also stores V transposed: Vt[b][g][j][k]  (k contiguous for the PV pass).
// One block per token row (B*S = 1024 blocks), 128 threads.
// ---------------------------------------------------------------------------
__global__ __launch_bounds__(128)
void SA_prep_kernel(const float* __restrict__ x,
                    const float* __restrict__ Wq,
                    const float* __restrict__ Wk,
                    const float* __restrict__ Wv,
                    const float* __restrict__ W1,
                    const float* __restrict__ b1,
                    unsigned short* __restrict__ qw1,    // bf16 [B*S][D]
                    unsigned short* __restrict__ nkw1,   // bf16 [B*S][D]
                    float* __restrict__ Vt)              // f32  [B][NG][DG][S]
{
  const int row = blockIdx.x;       // 0..B*S-1
  const int t   = threadIdx.x;      // 0..127
  __shared__ float xs[D_], Qs[D_], Ks[D_];
  xs[t] = x[(size_t)row * D_ + t];
  __syncthreads();
  float q = 0.f, k = 0.f, v = 0.f;
  #pragma unroll 8
  for (int i = 0; i < D_; ++i) {
    const float xv = xs[i];
    q = fmaf(xv, Wq[i * D_ + t], q);
    k = fmaf(xv, Wk[i * D_ + t], k);
    v = fmaf(xv, Wv[i * D_ + t], v);
  }
  // V column t of this row is Vr[b,k, j=t>>1, g=t&1] (reference reshape (DG,NG))
  {
    const int bb = row >> 9, kk = row & (S_ - 1);
    const int jj = t >> 1, gg = t & 1;
    Vt[(size_t)((bb * NG_ + gg) * DG_ + jj) * S_ + kk] = v;
  }
  Qs[t] = q; Ks[t] = k;
  __syncthreads();
  // per-group layer-1: out[c] = sum_i Q[g*64+i] * W1[i*64+c] (+ b1 on Q side)
  const int g = t >> 6, c = t & 63;
  float aq = b1[c], ak = 0.f;
  #pragma unroll 8
  for (int i = 0; i < DG_; ++i) {
    const float w = W1[i * DG_ + c];
    aq = fmaf(Qs[g * DG_ + i], w, aq);
    ak = fmaf(Ks[g * DG_ + i], w, ak);
  }
  qw1[(size_t)row * D_ + t]  = f2bfu(aq);
  nkw1[(size_t)row * D_ + t] = f2bfu(-ak);
}

// Build 32x16 bf16 B fragment (hT) for one K-chunk, fully packed bf16.
// Layout used here (16-bit B 32x16, contiguous k, same as A): lane<16 -> col = lane, rows K = 0..15;
// lane>=16 -> col = lane-16, rows K = 16..31; element j -> row (chunk*32 + half*16 + j)
// => per lane: 16 CONTIGUOUS bf16 of this key's (negated) KW1 group row.
__device__ __forceinline__ v16bf make_bfrag(const unsigned short* __restrict__ krow,
                                            int off, uint4 qa, uint4 qb) {
  const uint4 ka = *(const uint4*)(krow + off);
  const uint4 kb = *(const uint4*)(krow + off + 8);
  union { unsigned int u[8]; v16bf v; } r;
  r.u[0] = pk_relu_add(qa.x, ka.x);
  r.u[1] = pk_relu_add(qa.y, ka.y);
  r.u[2] = pk_relu_add(qa.z, ka.z);
  r.u[3] = pk_relu_add(qa.w, ka.w);
  r.u[4] = pk_relu_add(qb.x, kb.x);
  r.u[5] = pk_relu_add(qb.y, kb.y);
  r.u[6] = pk_relu_add(qb.z, kb.z);
  r.u[7] = pk_relu_add(qb.w, kb.w);
  return r.v;
}

// ---------------------------------------------------------------------------
// Kernel 2: per-(b, group, query) scoring MLP (WMMA), softmax, attention*V.
// WMMA computes D = W2^T (16x64) x h^T (64x16keys) + b2 (as C operand):
//   D[n, key] -> lane<16 holds key=lane, n=0..7; lane>=16 holds same key, n=8..15.
// Layer-3 reduction = 8 in-lane FMAs + ONE v_permlanex16.
// Grid = B*NG*S = 2048 blocks, 256 threads (8 wave32s).
// ---------------------------------------------------------------------------
__global__ __launch_bounds__(256)
void SA_attn_kernel(const unsigned short* __restrict__ qw1,
                    const unsigned short* __restrict__ nkw1,
                    const float* __restrict__ Vt,       // [B][NG][DG][S]
                    const unsigned short* __restrict__ afw, // bf16 [32][2][16]
                    const float* __restrict__ addB,     // [1,NG,1,S,S]
                    const float* __restrict__ multB,    // [1,NG,1,S,S]
                    const float* __restrict__ b2,       // [16]
                    const float* __restrict__ W3,       // [16,1]
                    const float* __restrict__ b3,       // [1]
                    float* __restrict__ out)            // [B,S,D]
{
  const int tid  = threadIdx.x;
  const int lane = tid & 31;
  const int wave = tid >> 5;
  const int idx  = blockIdx.x;
  const int q = idx & (S_ - 1);
  const int g = (idx >> 9) & (NG_ - 1);
  const int b = idx >> 10;

  const int r15  = lane & 15;   // A-row (n) / B-col (key) role
  const int half = lane >> 4;

  __shared__ float scores[S_];
  __shared__ float red[256];

  // Pre-packed W2^T A fragments (2 x 32-byte vector loads).
  const v16bf af0 = *(const v16bf*)(afw + lane * 32);
  const v16bf af1 = *(const v16bf*)(afw + lane * 32 + 16);

  // C operand = b2 broadcast: C[n,key] = b2[n], n = r + 8*half. Layer-3 weights.
  v8f accInit;
  float w3h[8];
  #pragma unroll
  for (int r = 0; r < 8; ++r) {
    accInit[r] = b2[r + 8 * half];
    w3h[r]     = W3[r + 8 * half];
  }
  const float b3v = b3[0];

  // Query-side layer-1 row, kept packed bf16 (same contiguous slices as B rows).
  const unsigned short* qrow = qw1 + (size_t)(b * S_ + q) * D_ + g * DG_;
  const uint4 q0a = *(const uint4*)(qrow + half * 16);
  const uint4 q0b = *(const uint4*)(qrow + half * 16 + 8);
  const uint4 q1a = *(const uint4*)(qrow + 32 + half * 16);
  const uint4 q1b = *(const uint4*)(qrow + 32 + half * 16 + 8);

  const unsigned short* kbase = nkw1 + (size_t)(b * S_) * D_ + g * DG_;
  const float* multP = multB + ((size_t)g * S_ + q) * S_;
  const float* addP  = addB  + ((size_t)g * S_ + q) * S_;

  // 512 keys / (8 waves * 16-key tiles) = 4 tiles per wave.
  for (int t = 0; t < 4; ++t) {
    const int tileBase = (wave * 4 + t) * 16;
    const unsigned short* krow = kbase + (size_t)(tileBase + r15) * D_;

    const v16bf bf0 = make_bfrag(krow, half * 16,      q0a, q0b);
    const v16bf bf1 = make_bfrag(krow, 32 + half * 16, q1a, q1b);
    v8f acc;
    acc = __builtin_amdgcn_wmma_f32_16x16x32_bf16(false, af0, false, bf0,
                                                  (short)0, accInit, false, false);
    acc = __builtin_amdgcn_wmma_f32_16x16x32_bf16(false, af1, false, bf1,
                                                  (short)0, acc, false, false);

    // layer-2 relu + layer-3 partial dot (n = r + 8*half), all in-lane
    float ts = 0.f;
    #pragma unroll
    for (int r = 0; r < 8; ++r)
      ts = fmaf(reluf(acc[r]), w3h[r], ts);

    // merge n=0..7 (lanes 0-15) with n=8..15 (lanes 16-31) for the same key
    ts += xrow_swap(ts);

    if (half == 0) {   // 16 writer lanes, one key each
      const int kk = tileBase + r15;
      const float av = reluf(ts + b3v);
      scores[kk] = av * multP[kk] + addP[kk];
    }
  }
  __syncthreads();

  // ---- softmax over 512 keys ----
  float lmax = -3.0e38f;
  for (int k2 = tid; k2 < S_; k2 += 256) lmax = fmaxf(lmax, scores[k2]);
  red[tid] = lmax; __syncthreads();
  for (int st = 128; st > 0; st >>= 1) {
    if (tid < st) red[tid] = fmaxf(red[tid], red[tid + st]);
    __syncthreads();
  }
  const float gmax = red[0];
  __syncthreads();
  float lsum = 0.f;
  for (int k2 = tid; k2 < S_; k2 += 256) {
    const float e = __expf(scores[k2] - gmax);
    scores[k2] = e;
    lsum += e;
  }
  red[tid] = lsum; __syncthreads();
  for (int st = 128; st > 0; st >>= 1) {
    if (tid < st) red[tid] += red[tid + st];
    __syncthreads();
  }
  const float inv = 1.f / red[0];
  __syncthreads();

  // ---- attention * V: out[b,q,j,g] = inv * sum_k p[k] * Vt[b,g,j,k] ----
  const int j = tid >> 2;          // 0..63
  const int part = tid & 3;        // 4-way key split, 128 keys each
  const float4* vp = (const float4*)(Vt +
      (size_t)((b * NG_ + g) * DG_ + j) * S_ + part * 128);
  float acc = 0.f;
  #pragma unroll 4
  for (int i = 0; i < 32; ++i) {
    const float4 v4 = vp[i];
    const int kb = part * 128 + i * 4;
    acc = fmaf(scores[kb + 0], v4.x, acc);
    acc = fmaf(scores[kb + 1], v4.y, acc);
    acc = fmaf(scores[kb + 2], v4.z, acc);
    acc = fmaf(scores[kb + 3], v4.w, acc);
  }
  red[tid] = acc;
  __syncthreads();
  if (tid < 64) {
    const float o = red[tid * 4 + 0] + red[tid * 4 + 1] +
                    red[tid * 4 + 2] + red[tid * 4 + 3];
    out[(size_t)(b * S_ + q) * D_ + tid * NG_ + g] = o * inv;
  }
}

// ---------------------------------------------------------------------------
// Launch. Input order: x, add_bias, mult_bias, Wq, Wk, Wv, W1, b1, W2, b2, W3, b3
// Workspace layout (needs 1 MB + 2 KB):
//   [0 , 256K)    qw1   bf16 [B*S][128]
//   [256K,512K)   nkw1  bf16 [B*S][128]   (negated K@W1)
//   [512K,1M)     Vt    f32  [B][NG][DG][S]
//   [1M, 1M+2K)   afw   bf16 [32][2][16]  (W2^T in WMMA A layout)
// ---------------------------------------------------------------------------
extern "C" void kernel_launch(void* const* d_in, const int* in_sizes, int n_in,
                              void* d_out, int out_size, void* d_ws, size_t ws_size,
                              hipStream_t stream) {
  const float* x     = (const float*)d_in[0];
  const float* addB  = (const float*)d_in[1];
  const float* multB = (const float*)d_in[2];
  const float* Wq    = (const float*)d_in[3];
  const float* Wk    = (const float*)d_in[4];
  const float* Wv    = (const float*)d_in[5];
  const float* W1    = (const float*)d_in[6];
  const float* b1    = (const float*)d_in[7];
  const float* W2    = (const float*)d_in[8];
  const float* b2    = (const float*)d_in[9];
  const float* W3    = (const float*)d_in[10];
  const float* b3    = (const float*)d_in[11];

  unsigned short* qw1  = (unsigned short*)d_ws;
  unsigned short* nkw1 = qw1 + (size_t)B_ * S_ * D_;
  float*          Vt   = (float*)(nkw1 + (size_t)B_ * S_ * D_);
  unsigned short* afw  = (unsigned short*)(Vt + (size_t)B_ * NG_ * DG_ * S_);
  float*          out  = (float*)d_out;

  SA_w2_kernel<<<1, 32, 0, stream>>>(W2, afw);
  SA_prep_kernel<<<B_ * S_, 128, 0, stream>>>(x, Wq, Wk, Wv, W1, b1,
                                              qw1, nkw1, Vt);
  SA_attn_kernel<<<B_ * NG_ * S_, 256, 0, stream>>>(qw1, nkw1, Vt, afw,
                                                    addB, multB,
                                                    b2, W3, b3, out);
}
